// BitNetMLP_4140348473395
// MI455X (gfx1250) — matrix operand order repaired and checked by reading, not compiled, not run on hardware
//
#include <hip/hip_runtime.h>
#include <hip/hip_bf16.h>

typedef _Float16 h4  __attribute__((ext_vector_type(4)));
typedef _Float16 h8  __attribute__((ext_vector_type(8)));
typedef _Float16 v16h __attribute__((ext_vector_type(16)));
typedef float    v8f  __attribute__((ext_vector_type(8)));
typedef int      v4i  __attribute__((ext_vector_type(4)));

#define HIDDEN 4096
#define INTER  11008
#define MTOT   4096

#define BM 128
#define BN 128
#define BK 64
#define LDP 72   // LDS row pitch in halves (144B: 16B-aligned, conflict-breaking)

#if defined(__has_builtin)
#if __has_builtin(__builtin_amdgcn_global_load_async_to_lds_b128)
#define HAVE_ASYNC_B128 1
#endif
#if __has_builtin(__builtin_amdgcn_s_wait_asynccnt)
#define HAVE_WAIT_ASYNC 1
#endif
#endif

// Builtin parameter types (from clang diagnostic): v4i addrspace(1)* / addrspace(3)*
typedef __attribute__((address_space(1))) v4i* gvp_t;
typedef __attribute__((address_space(3))) v4i* lvp_t;

// Async copy 16B global -> LDS (ASYNCcnt-tracked). Generic-pointer low 32 bits
// are the LDS offset.
__device__ __forceinline__ void async_copy_b128(const void* g, void* l) {
#ifdef HAVE_ASYNC_B128
  __builtin_amdgcn_global_load_async_to_lds_b128(
      (gvp_t)(unsigned long long)(size_t)g,
      (lvp_t)(unsigned)(size_t)l, 0, 0);
#else
  unsigned lo = (unsigned)(size_t)l;
  unsigned long long ga = (unsigned long long)(size_t)g;
  asm volatile("global_load_async_to_lds_b128 %0, %1, off" :: "v"(lo), "v"(ga) : "memory");
#endif
}

__device__ __forceinline__ void wait_async0() {
#ifdef HAVE_WAIT_ASYNC
  __builtin_amdgcn_s_wait_asynccnt(0);
#else
  asm volatile("s_wait_asynccnt 0x0" ::: "memory");
#endif
}

__device__ __forceinline__ v16h make_v16(h8 lo, h8 hi) {
  union { v16h v; h8 h[2]; } u;
  u.h[0] = lo; u.h[1] = hi;
  return u.v;
}

__device__ __forceinline__ v8f zero8() {
  v8f z;
#pragma unroll
  for (int i = 0; i < 8; ++i) z[i] = 0.f;
  return z;
}

// ---------------------------------------------------------------------------
// Kernel 1: inter[m,n] = silu(x@gate_w^T * sg) * (x@up_w^T * su), f16 output
// ---------------------------------------------------------------------------
__global__ __launch_bounds__(256) void gate_up_silu_kernel(
    const float* __restrict__ x, const float* __restrict__ gw,
    const float* __restrict__ uw, const float* __restrict__ sg_p,
    const float* __restrict__ su_p, _Float16* __restrict__ inter)
{
  __shared__ _Float16 sA[BM * LDP];
  __shared__ _Float16 sG[BN * LDP];
  __shared__ _Float16 sU[BN * LDP];

  const int tid   = threadIdx.x;
  const int lane  = tid & 31;
  const int wave  = tid >> 5;     // 0..7
  const int waveM = wave >> 2;    // 0..1 -> 64 rows each
  const int waveN = wave & 3;     // 0..3 -> 32 cols each
  const int m0 = blockIdx.y * BM;
  const int n0 = blockIdx.x * BN;

  v8f accG[4][2], accU[4][2];
#pragma unroll
  for (int i = 0; i < 4; ++i)
#pragma unroll
    for (int j = 0; j < 2; ++j) { accG[i][j] = zero8(); accU[i][j] = zero8(); }

  const int lrow  = tid >> 4;   // 0..15
  const int lcol4 = tid & 15;   // float4 index within 64-wide K slab

  // One base pointer per stream; row deltas (r*HIDDEN*4 <= 1.75MB) fold into
  // the 24-bit instruction offset, so each iteration does 3 u64 adds total.
  const float* pX = x  + (size_t)(m0 + lrow) * HIDDEN + lcol4 * 4;
  const float* pG = gw + (size_t)(n0 + lrow) * HIDDEN + lcol4 * 4;
  const float* pU = uw + (size_t)(n0 + lrow) * HIDDEN + lcol4 * 4;

  for (int kb = 0; kb < HIDDEN; kb += BK) {
    __syncthreads();
    // stage A tile (x), converting f32 -> f16
#pragma unroll
    for (int r = 0; r < BM; r += 16) {
      float4 v = *(const float4*)(pX + r * HIDDEN);
      h4 h = { (_Float16)v.x, (_Float16)v.y, (_Float16)v.z, (_Float16)v.w };
      *(h4*)(sA + (r + lrow) * LDP + lcol4 * 4) = h;
    }
    // stage gate & up weight tiles (rows of W are K-contiguous)
#pragma unroll
    for (int r = 0; r < BN; r += 16) {
      float4 g = *(const float4*)(pG + r * HIDDEN);
      float4 u = *(const float4*)(pU + r * HIDDEN);
      h4 hg = { (_Float16)g.x, (_Float16)g.y, (_Float16)g.z, (_Float16)g.w };
      h4 hu = { (_Float16)u.x, (_Float16)u.y, (_Float16)u.z, (_Float16)u.w };
      *(h4*)(sG + (r + lrow) * LDP + lcol4 * 4) = hg;
      *(h4*)(sU + (r + lrow) * LDP + lcol4 * 4) = hu;
    }
    __syncthreads();

    // Prefetch next K-slab (global_prefetch_b8) while the WMMAs below run;
    // +BK and r offsets fold into instruction immediates.
    if (kb + BK < HIDDEN) {
#pragma unroll
      for (int r = 0; r < BM; r += 16)
        __builtin_prefetch(pX + r * HIDDEN + BK, 0, 3);
#pragma unroll
      for (int r = 0; r < BN; r += 16) {
        __builtin_prefetch(pG + r * HIDDEN + BK, 0, 3);
        __builtin_prefetch(pU + r * HIDDEN + BK, 0, 3);
      }
    }

#pragma unroll
    for (int ks = 0; ks < BK; ks += 32) {
      // A fragments: lane<16 holds K 0..15, lane>=16 holds K 16..31
      const int kA = ks + ((lane < 16) ? 0 : 16);
      v16h a[4];
#pragma unroll
      for (int mi = 0; mi < 4; ++mi) {
        const _Float16* p = sA + (waveM * 64 + mi * 16 + (lane & 15)) * LDP;
        a[mi] = make_v16(*(const h8*)(p + kA), *(const h8*)(p + kA + 8));
      }
      // B fragments: lane<16 holds K 0..15 of column N=lane
      const int kB = ks + ((lane < 16) ? 0 : 16);
#pragma unroll
      for (int ni = 0; ni < 2; ++ni) {
        const _Float16* pg = sG + (waveN * 32 + ni * 16 + (lane & 15)) * LDP + kB;
        const _Float16* pu = sU + (waveN * 32 + ni * 16 + (lane & 15)) * LDP + kB;
        v16h bg = make_v16(*(const h8*)pg, *(const h8*)(pg + 8));
        v16h bu = make_v16(*(const h8*)pu, *(const h8*)(pu + 8));
#pragma unroll
        for (int mi = 0; mi < 4; ++mi) {
          accG[mi][ni] = __builtin_amdgcn_wmma_f32_16x16x32_f16(
              false, a[mi], false, bg, (short)0, accG[mi][ni], false, false);
          accU[mi][ni] = __builtin_amdgcn_wmma_f32_16x16x32_f16(
              false, a[mi], false, bu, (short)0, accU[mi][ni], false, false);
        }
      }
    }

    pX += BK; pG += BK; pU += BK;
  }

  // SwiGLU epilogue; D layout: VGPR j -> row j (lanes 0-15) / j+8 (lanes 16-31)
  const float sg = *sg_p, su = *su_p;
  const int mw = m0 + waveM * 64 + ((lane < 16) ? 0 : 8);
  const int nw = n0 + waveN * 32 + (lane & 15);
  _Float16* po = inter + (size_t)mw * INTER + nw;   // row deltas become immediates
#pragma unroll
  for (int mi = 0; mi < 4; ++mi)
#pragma unroll
    for (int ni = 0; ni < 2; ++ni)
#pragma unroll
      for (int j = 0; j < 8; ++j) {
        float g = accG[mi][ni][j] * sg;
        float u = accU[mi][ni][j] * su;
        float v = (g / (1.f + __expf(-g))) * u;
        po[(mi * 16 + j) * INTER + ni * 16] = (_Float16)v;
      }
}

// ---------------------------------------------------------------------------
// Kernel 2: out[m,h] = (inter @ down_w^T) * sd, f32 output.
// Double-buffered, single barrier per K-step:
//   A tile (f16) via GLOBAL_LOAD_ASYNC_TO_LDS_B128 (ASYNCcnt pipeline)
//   B tile (f32->f16) via register prefetch issued after the barrier
// ---------------------------------------------------------------------------
__global__ __launch_bounds__(256) void down_proj_kernel(
    const _Float16* __restrict__ inter, const float* __restrict__ dw,
    const float* __restrict__ sd_p, float* __restrict__ out)
{
  __shared__ _Float16 sA[2][BM * LDP];
  __shared__ _Float16 sB[2][BN * LDP];

  const int tid   = threadIdx.x;
  const int lane  = tid & 31;
  const int wave  = tid >> 5;
  const int waveM = wave >> 2;
  const int waveN = wave & 3;
  const int m0 = blockIdx.y * BM;
  const int n0 = blockIdx.x * BN;

  v8f acc[4][2];
#pragma unroll
  for (int i = 0; i < 4; ++i)
#pragma unroll
    for (int j = 0; j < 2; ++j) acc[i][j] = zero8();

  const int arow   = tid >> 3;  // 0..31; A tile: 8x 16B chunks per 64-half row
  const int achunk = tid & 7;
  const int lrow   = tid >> 4;  // B staging
  const int lcol4  = tid & 15;

  // Single advancing base pointer per stream (row deltas -> immediates for B;
  // async keeps explicit addresses since its IOFFSET shifts LDS + global alike).
  const _Float16* pA = inter + (size_t)(m0 + arow) * INTER + achunk * 8;
  const float*    pB = dw    + (size_t)(n0 + lrow) * INTER + lcol4 * 4;

  float4 breg[8];

  // ---- prologue: async A tile 0 in flight, B tile 0 into registers
#pragma unroll
  for (int r = 0; r < BM; r += 32)
    async_copy_b128(pA + r * INTER, &sA[0][(r + arow) * LDP + achunk * 8]);
#pragma unroll
  for (int r8 = 0; r8 < 8; ++r8)
    breg[r8] = *(const float4*)(pB + r8 * 16 * INTER);
  pA += BK; pB += BK;

  const int nK = INTER / BK;   // 172
  for (int i = 0; i < nK; ++i) {
    const int cur = i & 1;

    // commit staged B registers into sB[cur] (f32 -> f16)
#pragma unroll
    for (int r8 = 0; r8 < 8; ++r8) {
      float4 b = breg[r8];
      h4 hb = { (_Float16)b.x, (_Float16)b.y, (_Float16)b.z, (_Float16)b.w };
      *(h4*)(&sB[cur][(r8 * 16 + lrow) * LDP + lcol4 * 4]) = hb;
    }
    wait_async0();     // this wave's async chunks for sA[cur] have landed
    __syncthreads();   // everyone's sA[cur]/sB[cur] complete & visible

    // kick off next tile: async A -> sA[cur^1], B global loads -> breg.
    // Safe: sA[cur^1]/breg were last consumed before the barrier above.
    if (i + 1 < nK) {
#pragma unroll
      for (int r = 0; r < BM; r += 32)
        async_copy_b128(pA + r * INTER, &sA[cur ^ 1][(r + arow) * LDP + achunk * 8]);
#pragma unroll
      for (int r8 = 0; r8 < 8; ++r8)
        breg[r8] = *(const float4*)(pB + r8 * 16 * INTER);
      pA += BK; pB += BK;
    }

    // compute on current buffers while next-tile traffic is in flight
#pragma unroll
    for (int ks = 0; ks < BK; ks += 32) {
      const int kA = ks + ((lane < 16) ? 0 : 16);
      v16h a[4];
#pragma unroll
      for (int mi = 0; mi < 4; ++mi) {
        const _Float16* p = &sA[cur][(waveM * 64 + mi * 16 + (lane & 15)) * LDP];
        a[mi] = make_v16(*(const h8*)(p + kA), *(const h8*)(p + kA + 8));
      }
      const int kB = ks + ((lane < 16) ? 0 : 16);
#pragma unroll
      for (int ni = 0; ni < 2; ++ni) {
        const _Float16* pb = &sB[cur][(waveN * 32 + ni * 16 + (lane & 15)) * LDP] + kB;
        v16h b = make_v16(*(const h8*)pb, *(const h8*)(pb + 8));
#pragma unroll
        for (int mi = 0; mi < 4; ++mi) {
          acc[mi][ni] = __builtin_amdgcn_wmma_f32_16x16x32_f16(
              false, a[mi], false, b, (short)0, acc[mi][ni], false, false);
        }
      }
    }
  }

  const float sd = *sd_p;
  const int mw = m0 + waveM * 64 + ((lane < 16) ? 0 : 8);
  const int nw = n0 + waveN * 32 + (lane & 15);
  float* po = out + (size_t)mw * HIDDEN + nw;   // row deltas become immediates
#pragma unroll
  for (int mi = 0; mi < 4; ++mi)
#pragma unroll
    for (int ni = 0; ni < 2; ++ni)
#pragma unroll
      for (int j = 0; j < 8; ++j)
        po[(mi * 16 + j) * HIDDEN + ni * 16] = acc[mi][ni][j] * sd;
}

// ---------------------------------------------------------------------------
extern "C" void kernel_launch(void* const* d_in, const int* in_sizes, int n_in,
                              void* d_out, int out_size, void* d_ws, size_t ws_size,
                              hipStream_t stream) {
  const float* x  = (const float*)d_in[0];
  const float* gw = (const float*)d_in[1];
  const float* uw = (const float*)d_in[2];
  const float* dw = (const float*)d_in[3];
  const float* sg = (const float*)d_in[4];
  const float* su = (const float*)d_in[5];
  const float* sd = (const float*)d_in[6];

  _Float16* inter = (_Float16*)d_ws;   // MTOT x INTER f16 = ~90 MB scratch
  float* outp = (float*)d_out;

  dim3 blk(256);
  dim3 grid1(INTER / BN, MTOT / BM);   // 86 x 32
  dim3 grid2(HIDDEN / BN, MTOT / BM);  // 32 x 32

  gate_up_silu_kernel<<<grid1, blk, 0, stream>>>(x, gw, uw, sg, su, inter);
  down_proj_kernel<<<grid2, blk, 0, stream>>>(inter, dw, sd, outp);
}
